// DNALayer_39041252721353
// MI455X (gfx1250) — matrix operand order repaired and checked by reading, not compiled, not run on hardware
//
#include <hip/hip_runtime.h>
#include <hip/hip_bf16.h>
#include <math.h>

typedef __attribute__((ext_vector_type(16))) __bf16 bf16x16;
typedef __attribute__((ext_vector_type(8)))  __bf16 bf16x8;
typedef __attribute__((ext_vector_type(8)))  float  f32x8;
typedef __attribute__((ext_vector_type(4)))  int    i32x4;

#define D_MODEL 768
#define SEQ     8192
#define BATCH   4
#define NHEAD   12
#define DHEAD   64
#define WIN     128
#define NBUK    512
#define NWIN    (SEQ / WIN)           // 64
#define BSROWS  (BATCH * SEQ)         // 32768
#define TILE_K  32
#define LSTR    40                    // padded LDS stride (elems): 80B, 16B-aligned, conflict-free
#define PSTR    136                   // padded P-strip stride (elems): 272B

// CDNA5 async global->LDS staging (ASYNCcnt-tracked), with safe fallback.
#if defined(__AMDGCN__) && __has_builtin(__builtin_amdgcn_global_load_async_to_lds_b128) && __has_builtin(__builtin_amdgcn_s_wait_asynccnt)
#define HAVE_ASYNC_LDS 1
#else
#define HAVE_ASYNC_LDS 0
#endif

// ---------------------------------------------------------------------------
// WMMA helper: D = A(16x32 bf16) x B(32x16 bf16) + C(16x16 f32)
// ---------------------------------------------------------------------------
__device__ __forceinline__ f32x8 wmma_bf16(bf16x16 a, bf16x16 b, f32x8 c) {
  return __builtin_amdgcn_wmma_f32_16x16x32_bf16(false, a, false, b,
                                                 (short)0, c, false, false);
}

__device__ __forceinline__ bf16x16 frag_cat(bf16x8 lo, bf16x8 hi) {
  return __builtin_shufflevector(lo, hi, 0, 1, 2, 3, 4, 5, 6, 7,
                                 8, 9, 10, 11, 12, 13, 14, 15);
}
// 16 contiguous bf16 (two 16B vector loads)
__device__ __forceinline__ bf16x16 frag_ld(const __bf16* p) {
  return frag_cat(*(const bf16x8*)p, *(const bf16x8*)(p + 8));
}
// A-matrix 16x32 fragment: lane half hh reads
// [hh*16 .. hh*16+15] from its 32-elem row (same k order as B fragments).
__device__ __forceinline__ bf16x16 frag_ld_A(const __bf16* row, int hh) {
  return frag_cat(*(const bf16x8*)(row + hh * 16),
                  *(const bf16x8*)(row + hh * 16 + 8));
}

// ---------------------------------------------------------------------------
// Convert f32 -> bf16
// ---------------------------------------------------------------------------
__global__ void f2b_kernel(const float* __restrict__ in, __bf16* __restrict__ out,
                           size_t n) {
  size_t i = (size_t)blockIdx.x * blockDim.x + threadIdx.x;
  if (i < n) out[i] = (__bf16)in[i];
}

// ---------------------------------------------------------------------------
// Transpose V: [B,S,D] bf16 -> [B,H,DH,S] bf16
// ---------------------------------------------------------------------------
__global__ void vtrans_kernel(const __bf16* __restrict__ v, __bf16* __restrict__ vt) {
  size_t i = (size_t)blockIdx.x * blockDim.x + threadIdx.x;
  if (i >= (size_t)BSROWS * D_MODEL) return;
  int s = (int)(i & (SEQ - 1));
  size_t r = i >> 13;                    // b*H*DH + h*DH + d
  int d = (int)(r % DHEAD);
  int h = (int)((r / DHEAD) % NHEAD);
  int b = (int)(r / (DHEAD * NHEAD));
  vt[i] = v[((size_t)b * SEQ + s) * D_MODEL + h * DHEAD + d];
}

// ---------------------------------------------------------------------------
// Generic bf16 WMMA GEMM:  C[M,N] = act(A[M,K] * W[K,N] + bias)
// Block 256 thr (8 waves); block tile 128x128; wave tile 16x128; K-step 32.
// W tile staged TRANSPOSED into B-fragment order: lB[col*LSTR + k].
// A tile staged via async global->LDS copy when available.
// ---------------------------------------------------------------------------
__global__ __launch_bounds__(256) void gemm_bias_kernel(
    const __bf16* __restrict__ A, const __bf16* __restrict__ W,
    const float* __restrict__ bias,
    float* __restrict__ outF, __bf16* __restrict__ outB,
    int M, int N, int K, int gelu) {
  __shared__ __bf16 lA[128 * LSTR];
  __shared__ __bf16 lB[128 * LSTR];

  const int tid  = threadIdx.x;
  const int wave = tid >> 5;
  const int lane = tid & 31;
  const int mm   = lane & 15;
  const int hh   = lane >> 4;
  const int row0 = blockIdx.x * 128;
  const int col0 = blockIdx.y * 128;

  // per-thread staging coordinates
  const int ar  = tid >> 1, aseg = tid & 1;              // A: row, 16-elem segment
  const int bcol = tid & 127, bkh = tid >> 7;            // W: column, k-half

  f32x8 acc[8] = {};

  for (int k0 = 0; k0 < K; k0 += TILE_K) {
    const __bf16* asrc = A + (size_t)(row0 + ar) * K + k0 + aseg * 16;
    __bf16* adst = lA + ar * LSTR + aseg * 16;
    { // stage A tile 128x32
#if HAVE_ASYNC_LDS
      __builtin_amdgcn_global_load_async_to_lds_b128(
          (i32x4*)asrc,
          (__attribute__((address_space(3))) i32x4*)adst, 0, 0);
      __builtin_amdgcn_global_load_async_to_lds_b128(
          (i32x4*)(asrc + 8),
          (__attribute__((address_space(3))) i32x4*)(adst + 8), 0, 0);
#else
      bf16x8 a0 = *(const bf16x8*)asrc;
      bf16x8 a1 = *(const bf16x8*)(asrc + 8);
      *(bf16x8*)adst = a0;
      *(bf16x8*)(adst + 8) = a1;
#endif
    }
    { // stage W tile 32x128 TRANSPOSED: thread gathers one column's 16 k-values
      const __bf16* src = W + (size_t)(k0 + bkh * 16) * N + col0 + bcol;
      bf16x8 v0, v1;
#pragma unroll
      for (int e = 0; e < 8; e++) v0[e] = src[(size_t)e * N];
#pragma unroll
      for (int e = 0; e < 8; e++) v1[e] = src[(size_t)(e + 8) * N];
      __bf16* dst = lB + bcol * LSTR + bkh * 16;
      *(bf16x8*)dst = v0;
      *(bf16x8*)(dst + 8) = v1;
    }
    // prefetch next K-step tiles while this tile's math runs
    if (k0 + TILE_K < K) {
      __builtin_prefetch(asrc + TILE_K, 0, 1);
      __builtin_prefetch(W + (size_t)(k0 + TILE_K + bkh * 16) * N + col0 + bcol, 0, 1);
    }
#if HAVE_ASYNC_LDS
    __builtin_amdgcn_s_wait_asynccnt(0);
#endif
    __syncthreads();

    bf16x16 af = frag_ld_A(lA + (wave * 16 + mm) * LSTR, hh);
#pragma unroll
    for (int t = 0; t < 8; t++) {
      bf16x16 bfr = frag_ld(lB + (t * 16 + mm) * LSTR + hh * 16);
      acc[t] = wmma_bf16(af, bfr, acc[t]);
    }
    __syncthreads();
  }

  // epilogue: C layout row = j + 8*hh, col = t*16 + mm
#pragma unroll
  for (int t = 0; t < 8; t++) {
    int col = col0 + t * 16 + mm;
    float bv = bias[col];
#pragma unroll
    for (int j = 0; j < 8; j++) {
      int row = row0 + wave * 16 + j + 8 * hh;
      float v = acc[t][j] + bv;
      if (gelu) v = 0.5f * v * (1.0f + erff(v * 0.70710678118654752f));
      size_t idx = (size_t)row * N + col;
      if (outF) outF[idx] = v;
      if (outB) outB[idx] = (__bf16)v;
    }
  }
}

// ---------------------------------------------------------------------------
// Local windowed causal attention (flash, single 128-key chunk).
// grid = B*H*NWIN, 256 thr (8 waves); wave handles 16 query rows.
// q/k bf16 [B,S,D]; vt bf16 [B,H,DH,S]; out bf16 [B,S,D].
// ---------------------------------------------------------------------------
__global__ __launch_bounds__(256) void local_attn_kernel(
    const __bf16* __restrict__ qb, const __bf16* __restrict__ kb,
    const __bf16* __restrict__ vt, __bf16* __restrict__ ob) {
  __shared__ __bf16 lP[8 * 16 * PSTR];  // per-wave P transpose strip

  const int tid  = threadIdx.x;
  const int wave = tid >> 5;
  const int lane = tid & 31;
  const int mm   = lane & 15;
  const int hh   = lane >> 4;

  int blk = blockIdx.x;
  int win = blk % NWIN;
  int h   = (blk / NWIN) % NHEAD;
  int b   = blk / (NWIN * NHEAD);
  const size_t base  = (size_t)b * SEQ * D_MODEL;
  const size_t vbase = ((size_t)b * NHEAD + h) * DHEAD * SEQ;
  const int s0 = win * WIN;
  const int fh = h * DHEAD;
  __bf16* myP = lP + wave * 16 * PSTR;

  // Q fragments (16 rows x 64 K, two 32-K steps)
  bf16x16 qa[2];
#pragma unroll
  for (int kk = 0; kk < 2; kk++)
    qa[kk] = frag_ld_A(qb + base + (size_t)(s0 + wave * 16 + mm) * D_MODEL + fh + kk * 32, hh);

  // scores: 16 x 128
  f32x8 cs[8] = {};
#pragma unroll
  for (int t = 0; t < 8; t++) {
#pragma unroll
    for (int kk = 0; kk < 2; kk++) {
      bf16x16 bk = frag_ld(kb + base + (size_t)(s0 + t * 16 + mm) * D_MODEL + fh + kk * 32 + hh * 16);
      cs[t] = wmma_bf16(qa[kk], bk, cs[t]);
    }
  }

  const float scale = 0.125f;  // 1/sqrt(64)
#pragma unroll
  for (int j = 0; j < 8; j++) {
    int row = wave * 16 + j + 8 * hh;  // query row within window
    float mx = -3.0e38f;
#pragma unroll
    for (int t = 0; t < 8; t++) {
      int col = t * 16 + mm;
      float v = cs[t][j] * scale;
      if (col > row) v = -3.0e38f;     // causal within window
      cs[t][j] = v;
      mx = fmaxf(mx, v);
    }
#pragma unroll
    for (int d = 1; d < 16; d <<= 1) mx = fmaxf(mx, __shfl_xor(mx, d, 32));
    float sm = 0.0f;
#pragma unroll
    for (int t = 0; t < 8; t++) {
      float p = __expf(cs[t][j] - mx);
      cs[t][j] = p;
      sm += p;
    }
#pragma unroll
    for (int d = 1; d < 16; d <<= 1) sm += __shfl_xor(sm, d, 32);
    float inv = 1.0f / sm;
#pragma unroll
    for (int t = 0; t < 8; t++)
      myP[(j + 8 * hh) * PSTR + t * 16 + mm] = (__bf16)(cs[t][j] * inv);
  }
  __syncthreads();

  // O = P(16x128) x V(128x64); V fragments contiguous from vt
#pragma unroll
  for (int nt = 0; nt < 4; nt++) {
    f32x8 o = {};
#pragma unroll
    for (int ks = 0; ks < 4; ks++) {
      bf16x16 pa = frag_ld_A(myP + mm * PSTR + ks * 32, hh);
      bf16x16 bv = frag_ld(vt + vbase + (size_t)(nt * 16 + mm) * SEQ + s0 + ks * 32 + hh * 16);
      o = wmma_bf16(pa, bv, o);
    }
#pragma unroll
    for (int j = 0; j < 8; j++) {
      int srow = s0 + wave * 16 + j + 8 * hh;
      ob[base + (size_t)srow * D_MODEL + fh + nt * 16 + mm] = (__bf16)o[j];
    }
  }
}

// ---------------------------------------------------------------------------
// Global bucket attention (flash over 4 chunks of 128 buckets, no mask).
// grid = B*H*(S/128), 256 thr; wave = 16 query rows.
// gk bf16 [H][NBUK][DH]; gvT bf16 [H][DH][NBUK].
// ---------------------------------------------------------------------------
__global__ __launch_bounds__(256) void global_attn_kernel(
    const __bf16* __restrict__ qb, const __bf16* __restrict__ gk,
    const __bf16* __restrict__ gvT, __bf16* __restrict__ ob) {
  __shared__ __bf16 lP[8 * 16 * PSTR];

  const int tid  = threadIdx.x;
  const int wave = tid >> 5;
  const int lane = tid & 31;
  const int mm   = lane & 15;
  const int hh   = lane >> 4;

  int blk = blockIdx.x;
  int st  = blk % (SEQ / 128);
  int h   = (blk / (SEQ / 128)) % NHEAD;
  int b   = blk / ((SEQ / 128) * NHEAD);
  const size_t base = (size_t)b * SEQ * D_MODEL;
  const int s0 = st * 128;
  const int fh = h * DHEAD;
  const size_t kbase = (size_t)h * NBUK * DHEAD;  // [H][NB][DH]
  const size_t vbase = (size_t)h * DHEAD * NBUK;  // [H][DH][NB]
  __bf16* myP = lP + wave * 16 * PSTR;

  bf16x16 qa[2];
#pragma unroll
  for (int kk = 0; kk < 2; kk++)
    qa[kk] = frag_ld_A(qb + base + (size_t)(s0 + wave * 16 + mm) * D_MODEL + fh + kk * 32, hh);

  float Mx[8], Ls[8];
  f32x8 o[4] = {};
#pragma unroll
  for (int j = 0; j < 8; j++) { Mx[j] = -3.0e38f; Ls[j] = 0.0f; }

  const float scale = 0.125f;
  for (int c = 0; c < 4; c++) {              // 4 chunks of 128 buckets
    f32x8 cs[8] = {};
#pragma unroll
    for (int t = 0; t < 8; t++) {
#pragma unroll
      for (int kk = 0; kk < 2; kk++) {
        bf16x16 bk = frag_ld(gk + kbase + (size_t)(c * 128 + t * 16 + mm) * DHEAD + kk * 32 + hh * 16);
        cs[t] = wmma_bf16(qa[kk], bk, cs[t]);
      }
    }

#pragma unroll
    for (int j = 0; j < 8; j++) {
      float mx = -3.0e38f;
#pragma unroll
      for (int t = 0; t < 8; t++) {
        float v = cs[t][j] * scale;
        cs[t][j] = v;
        mx = fmaxf(mx, v);
      }
#pragma unroll
      for (int d = 1; d < 16; d <<= 1) mx = fmaxf(mx, __shfl_xor(mx, d, 32));
      float newM = fmaxf(Mx[j], mx);
      float alpha = __expf(Mx[j] - newM);
      float sm = 0.0f;
#pragma unroll
      for (int t = 0; t < 8; t++) {
        float p = __expf(cs[t][j] - newM);
        cs[t][j] = p;
        sm += p;
      }
#pragma unroll
      for (int d = 1; d < 16; d <<= 1) sm += __shfl_xor(sm, d, 32);
      Ls[j] = Ls[j] * alpha + sm;
      Mx[j] = newM;
#pragma unroll
      for (int nt = 0; nt < 4; nt++) o[nt][j] *= alpha;
#pragma unroll
      for (int t = 0; t < 8; t++)
        myP[(j + 8 * hh) * PSTR + t * 16 + mm] = (__bf16)cs[t][j];
    }
    __syncthreads();

#pragma unroll
    for (int nt = 0; nt < 4; nt++) {
#pragma unroll
      for (int ks = 0; ks < 4; ks++) {
        bf16x16 pa = frag_ld_A(myP + mm * PSTR + ks * 32, hh);
        bf16x16 bv = frag_ld(gvT + vbase + (size_t)(nt * 16 + mm) * NBUK + c * 128 + ks * 32 + hh * 16);
        o[nt] = wmma_bf16(pa, bv, o[nt]);
      }
    }
    __syncthreads();
  }

#pragma unroll
  for (int nt = 0; nt < 4; nt++) {
#pragma unroll
    for (int j = 0; j < 8; j++) {
      int srow = s0 + wave * 16 + j + 8 * hh;
      ob[base + (size_t)srow * D_MODEL + fh + nt * 16 + mm] = (__bf16)(o[nt][j] / Ls[j]);
    }
  }
}

// ---------------------------------------------------------------------------
// y = LayerNorm(x + r) * g + b ; optional bf16 copy.  1 block per row.
// ---------------------------------------------------------------------------
__global__ __launch_bounds__(256) void residual_ln_kernel(
    const float* __restrict__ x, const float* __restrict__ r,
    const float* __restrict__ g, const float* __restrict__ be,
    float* __restrict__ y, __bf16* __restrict__ yb) {
  __shared__ float red0[256];
  __shared__ float red1[256];
  const int row = blockIdx.x;
  const int tid = threadIdx.x;
  const size_t off = (size_t)row * D_MODEL;

  float v[3];
  float s = 0.0f, s2 = 0.0f;
#pragma unroll
  for (int i = 0; i < 3; i++) {
    float t = x[off + tid + i * 256] + r[off + tid + i * 256];
    v[i] = t; s += t; s2 += t * t;
  }
  red0[tid] = s; red1[tid] = s2;
  __syncthreads();
  for (int o2 = 128; o2 > 0; o2 >>= 1) {
    if (tid < o2) { red0[tid] += red0[tid + o2]; red1[tid] += red1[tid + o2]; }
    __syncthreads();
  }
  float mean = red0[0] * (1.0f / D_MODEL);
  float var  = fmaxf(red1[0] * (1.0f / D_MODEL) - mean * mean, 0.0f);
  float inv  = rsqrtf(var + 1e-5f);
#pragma unroll
  for (int i = 0; i < 3; i++) {
    int c = tid + i * 256;
    float o = (v[i] - mean) * inv * g[c] + be[c];
    y[off + c] = o;
    if (yb) yb[off + c] = (__bf16)o;
  }
}

// ---------------------------------------------------------------------------
// Split kv [NB,2D] f32 -> gk bf16 [H][NB][DH], gvT bf16 [H][DH][NB]
// ---------------------------------------------------------------------------
__global__ void kv_split_kernel(const float* __restrict__ kv,
                                __bf16* __restrict__ gk, __bf16* __restrict__ gvT) {
  int i = blockIdx.x * blockDim.x + threadIdx.x;  // NB * D elements
  if (i >= NBUK * D_MODEL) return;
  int n = i / D_MODEL, f = i % D_MODEL;
  int h = f / DHEAD, d = f % DHEAD;
  gk[((size_t)h * NBUK + n) * DHEAD + d]  = (__bf16)kv[(size_t)n * (2 * D_MODEL) + f];
  gvT[((size_t)h * DHEAD + d) * NBUK + n] = (__bf16)kv[(size_t)n * (2 * D_MODEL) + D_MODEL + f];
}

// ---------------------------------------------------------------------------
// Host orchestration
// ---------------------------------------------------------------------------
extern "C" void kernel_launch(void* const* d_in, const int* in_sizes, int n_in,
                              void* d_out, int out_size, void* d_ws, size_t ws_size,
                              hipStream_t stream) {
  (void)in_sizes; (void)n_in; (void)out_size; (void)ws_size;
  const float* x    = (const float*)d_in[0];
  const float* bm   = (const float*)d_in[2];
  const float* lq_w = (const float*)d_in[3];  const float* lq_b = (const float*)d_in[4];
  const float* lk_w = (const float*)d_in[5];  const float* lk_b = (const float*)d_in[6];
  const float* lv_w = (const float*)d_in[7];  const float* lv_b = (const float*)d_in[8];
  const float* lo_w = (const float*)d_in[9];  const float* lo_b = (const float*)d_in[10];
  const float* gq_w = (const float*)d_in[11]; const float* gq_b = (const float*)d_in[12];
  const float* gkv_w= (const float*)d_in[13]; const float* gkv_b= (const float*)d_in[14];
  const float* go_w = (const float*)d_in[15]; const float* go_b = (const float*)d_in[16];
  const float* f1_w = (const float*)d_in[17]; const float* f1_b = (const float*)d_in[18];
  const float* f2_w = (const float*)d_in[19]; const float* f2_b = (const float*)d_in[20];
  const float* n1_g = (const float*)d_in[21]; const float* n1_b = (const float*)d_in[22];
  const float* n2_g = (const float*)d_in[23]; const float* n2_b = (const float*)d_in[24];
  const float* n3_g = (const float*)d_in[25]; const float* n3_b = (const float*)d_in[26];
  float* out = (float*)d_out;

  const size_t D = D_MODEL, D2 = D * D, BSD = (size_t)BSROWS * D;
  uint8_t* p = (uint8_t*)d_ws;
  auto carve = [&](size_t bytes) -> void* {
    void* r = (void*)p; p += (bytes + 255) & ~(size_t)255; return r;
  };
  // bf16 weights
  __bf16* wq  = (__bf16*)carve(D2 * 2);
  __bf16* wk  = (__bf16*)carve(D2 * 2);
  __bf16* wv  = (__bf16*)carve(D2 * 2);
  __bf16* wo  = (__bf16*)carve(D2 * 2);
  __bf16* wgq = (__bf16*)carve(D2 * 2);
  __bf16* wgo = (__bf16*)carve(D2 * 2);
  __bf16* wgkv= (__bf16*)carve(D * 2 * D * 2);
  __bf16* wf1 = (__bf16*)carve(D * 4 * D * 2);
  __bf16* wf2 = (__bf16*)carve(4 * D * D * 2);
  __bf16* bmb = (__bf16*)carve((size_t)NBUK * D * 2);
  // activations
  __bf16* xb  = (__bf16*)carve(BSD * 2);   // x bf16; reused as x1b later
  __bf16* qb_ = (__bf16*)carve(BSD * 2);   // reused as gq bf16
  __bf16* kb_ = (__bf16*)carve(BSD * 2);   // reused as global attn out bf16
  __bf16* vb_ = (__bf16*)carve(BSD * 2);   // reused as x2b
  __bf16* vt  = (__bf16*)carve(BSD * 2);   // transposed V [B,H,DH,S]
  __bf16* aob = (__bf16*)carve(BSD * 2);
  __bf16* hb  = (__bf16*)carve(BSD * 4 * 2);
  float*  t1  = (float*) carve(BSD * 4);
  float*  x1  = (float*) carve(BSD * 4);
  float*  x2  = (float*) carve(BSD * 4);
  float*  kvf = (float*) carve((size_t)NBUK * 2 * D * 4);
  __bf16* gk  = (__bf16*)carve((size_t)NHEAD * NBUK * DHEAD * 2);
  __bf16* gvT = (__bf16*)carve((size_t)NHEAD * NBUK * DHEAD * 2);

  auto cvt = [&](const float* src, __bf16* dst, size_t n) {
    f2b_kernel<<<dim3((unsigned)((n + 255) / 256)), dim3(256), 0, stream>>>(src, dst, n);
  };
  auto gemm = [&](const __bf16* A, const __bf16* W, const float* bias,
                  float* oF, __bf16* oB, int M, int N, int K, int gelu) {
    gemm_bias_kernel<<<dim3(M / 128, N / 128), dim3(256), 0, stream>>>(
        A, W, bias, oF, oB, M, N, K, gelu);
  };

  // weight / input conversion
  cvt(lq_w, wq, D2);  cvt(lk_w, wk, D2);  cvt(lv_w, wv, D2);  cvt(lo_w, wo, D2);
  cvt(gq_w, wgq, D2); cvt(go_w, wgo, D2); cvt(gkv_w, wgkv, D * 2 * D);
  cvt(f1_w, wf1, D * 4 * D); cvt(f2_w, wf2, 4 * D * D);
  cvt(bm, bmb, (size_t)NBUK * D);
  cvt(x, xb, BSD);

  // ---- stage 1: local attention ----
  gemm(xb, wq, lq_b, nullptr, qb_, BSROWS, D_MODEL, D_MODEL, 0);
  gemm(xb, wk, lk_b, nullptr, kb_, BSROWS, D_MODEL, D_MODEL, 0);
  gemm(xb, wv, lv_b, nullptr, vb_, BSROWS, D_MODEL, D_MODEL, 0);
  vtrans_kernel<<<dim3((unsigned)((BSD + 255) / 256)), dim3(256), 0, stream>>>(vb_, vt);
  local_attn_kernel<<<dim3(BATCH * NHEAD * NWIN), dim3(256), 0, stream>>>(qb_, kb_, vt, aob);
  gemm(aob, wo, lo_b, t1, nullptr, BSROWS, D_MODEL, D_MODEL, 0);
  __bf16* x1b = xb;  // xb dead; reuse
  residual_ln_kernel<<<dim3(BSROWS), dim3(256), 0, stream>>>(x, t1, n1_g, n1_b, x1, x1b);

  // ---- stage 2: global bucket attention ----
  __bf16* gqb = qb_;  // reuse
  __bf16* gob = kb_;  // reuse
  gemm(x1b, wgq, gq_b, nullptr, gqb, BSROWS, D_MODEL, D_MODEL, 0);
  gemm(bmb, wgkv, gkv_b, kvf, nullptr, NBUK, 2 * D_MODEL, D_MODEL, 0);
  kv_split_kernel<<<dim3((NBUK * D_MODEL + 255) / 256), dim3(256), 0, stream>>>(kvf, gk, gvT);
  global_attn_kernel<<<dim3(BATCH * NHEAD * (SEQ / 128)), dim3(256), 0, stream>>>(gqb, gk, gvT, gob);
  gemm(gob, wgo, go_b, t1, nullptr, BSROWS, D_MODEL, D_MODEL, 0);
  __bf16* x2b = vb_;  // reuse
  residual_ln_kernel<<<dim3(BSROWS), dim3(256), 0, stream>>>(x1, t1, n2_g, n2_b, x2, x2b);

  // ---- stage 3: FFN ----
  gemm(x2b, wf1, f1_b, nullptr, hb, BSROWS, 4 * D_MODEL, D_MODEL, 1);
  gemm(hb, wf2, f2_b, t1, nullptr, BSROWS, D_MODEL, 4 * D_MODEL, 0);
  residual_ln_kernel<<<dim3(BSROWS), dim3(256), 0, stream>>>(x2, t1, n3_g, n3_b, out, nullptr);
}
